// Chunk_76373108457890
// MI455X (gfx1250) — matrix operand order repaired and checked by reading, not compiled, not run on hardware
//
#include <hip/hip_runtime.h>
#include <hip/hip_bf16.h>
#include <math.h>

// ---------------------------------------------------------------------------
// GRU (2 layers) + FC for MI455X (gfx1250), bf16 WMMA w/ fp32 accumulate.
//   B=32, T=512, IN=256, H=1024, G=3H=3072, OUT=256
//   * packed bf16 WMMA fragments (one 32B contiguous chunk per lane)
//   * batched input-projection GEMMs: 2x2 tile blocking, compile-time K loop
//     fully unrolled so the scheduler pipelines loads across WMMA groups
//   * persistent recurrence kernel: per-block w_hh slice pinned in LDS
//     (128KB of 320KB), grid barrier between GEMM and gate phases,
//     next-step gx prefetched during gate math
// ---------------------------------------------------------------------------

typedef __attribute__((ext_vector_type(16))) __bf16 v16bf;
typedef __attribute__((ext_vector_type(8)))  float  v8f;

#define DEVINL __device__ __forceinline__

// problem dims
constexpr int Bm   = 32;
constexpr int Tm   = 512;
constexpr int INm  = 256;
constexpr int Hm   = 1024;
constexpr int Gm   = 3 * Hm;          // 3072
constexpr int OUTm = 256;

constexpr int NTG   = Gm / 16;        // 192 gate N-tiles
constexpr int KTH   = Hm / 32;        // 32 K-tiles over H
constexpr int KTI   = INm / 32;       // 8  K-tiles over IN
constexpr int MTSEQ = (Bm * Tm) / 16; // 1024 sequence M-tiles
constexpr int MTB   = Bm / 16;        // 2 batch M-tiles
constexpr int NTOUT = OUTm / 16;      // 16

constexpr int SCAN_BLOCKS = 48;       // 48 blocks * 8 waves = 384 waves = 2*192 tiles
constexpr int LDS_NT = 4;             // nt tiles per scan block (4*32 frags = 128KB)

// ---------------- workspace layout (bytes, all >=4KB aligned) ----------------
constexpr size_t OFF_GX    = 0;                                   // [B*T, G] f32
constexpr size_t SZ_GX     = (size_t)Bm * Tm * Gm * 4;            // 201326592
constexpr size_t OFF_SEQ   = OFF_GX + SZ_GX;                      // seq A-frags bf16
constexpr size_t SZ_SEQ    = (size_t)MTSEQ * KTH * 512 * 2;       // 33554432
constexpr size_t OFF_WIH0  = OFF_SEQ + SZ_SEQ;
constexpr size_t SZ_WIH0   = (size_t)NTG * KTI * 512 * 2;         // 1572864
constexpr size_t OFF_WHH0  = OFF_WIH0 + SZ_WIH0;
constexpr size_t SZ_WHH    = (size_t)NTG * KTH * 512 * 2;         // 6291456
constexpr size_t OFF_WIH1  = OFF_WHH0 + SZ_WHH;
constexpr size_t OFF_WHH1  = OFF_WIH1 + SZ_WHH;
constexpr size_t OFF_FCW   = OFF_WHH1 + SZ_WHH;
constexpr size_t SZ_FCW    = (size_t)NTOUT * KTH * 512 * 2;       // 524288
constexpr size_t OFF_H0    = OFF_FCW + SZ_FCW;                    // [B*H] f32
constexpr size_t OFF_H1    = OFF_H0 + (size_t)Bm * Hm * 4;
constexpr size_t OFF_HFRAG = OFF_H1 + (size_t)Bm * Hm * 4;        // h A-frags bf16
constexpr size_t SZ_HFRAG  = (size_t)MTB * KTH * 512 * 2;         // 65536
constexpr size_t OFF_GH    = OFF_HFRAG + SZ_HFRAG;                // [B, G] f32
constexpr size_t OFF_SYNC  = OFF_GH + (size_t)Bm * Gm * 4;        // barrier state

DEVINL unsigned short f2bf(float x) {
  union { float f; unsigned int u; } v; v.f = x;
  unsigned int r = v.u + 0x7FFFu + ((v.u >> 16) & 1u);   // round-to-nearest-even
  return (unsigned short)(r >> 16);
}

DEVINL v8f wmma_bf16(v16bf a, v16bf b, v8f c) {
  return __builtin_amdgcn_wmma_f32_16x16x32_bf16(false, a, false, b,
                                                 (short)0, c, false, false);
}

DEVINL float sigmoidf_(float x) { return 1.0f / (1.0f + __expf(-x)); }

// -------- grid-wide barrier (agent scope), generation-counter flavor --------
DEVINL void grid_sync(int* cnt, int* gen, int nblk, int target) {
  __syncthreads();
  if (threadIdx.x == 0) {
    int prev = __hip_atomic_fetch_add(cnt, 1, __ATOMIC_ACQ_REL, __HIP_MEMORY_SCOPE_AGENT);
    if (prev == nblk - 1) {
      __hip_atomic_store(cnt, 0, __ATOMIC_RELAXED, __HIP_MEMORY_SCOPE_AGENT);
      __hip_atomic_fetch_add(gen, 1, __ATOMIC_RELEASE, __HIP_MEMORY_SCOPE_AGENT);
    } else {
      while (__hip_atomic_load(gen, __ATOMIC_ACQUIRE, __HIP_MEMORY_SCOPE_AGENT) < target)
        __builtin_amdgcn_s_sleep(2);
    }
  }
  __syncthreads();
}

// ---------------------------------------------------------------------------
// Pack B-matrix fragments: B[k][n] = w[n][k], w is [Nrows, K] row-major.
// ---------------------------------------------------------------------------
__global__ void pack_w_kernel(const float* __restrict__ w,
                              unsigned short* __restrict__ out,
                              int NT, int KT, int K) {
  int idx = blockIdx.x * blockDim.x + threadIdx.x;
  if (idx >= NT * KT * 512) return;
  int within = idx & 511, fragId = idx >> 9;
  int lane = within >> 4, q = within & 15;
  int v = q >> 1, e = q & 1;
  int kt = fragId % KT, nt = fragId / KT;
  int n = nt * 16 + (lane & 15);
  int k = kt * 32 + ((v >= 4) ? 16 : 0) + ((lane >= 16) ? 8 : 0) + 2 * (v & 3) + e;
  out[idx] = f2bf(w[(size_t)n * K + k]);
}

// ---------------------------------------------------------------------------
// Pack A-matrix fragments from row-major activations x[M, K].
// ---------------------------------------------------------------------------
__global__ void pack_a_kernel(const float* __restrict__ x,
                              unsigned short* __restrict__ out,
                              int MT, int KT, int K) {
  int idx = blockIdx.x * blockDim.x + threadIdx.x;
  if (idx >= MT * KT * 512) return;
  int within = idx & 511, fragId = idx >> 9;
  int lane = within >> 4, q = within & 15;
  int v = q >> 1, e = q & 1;
  int kt = fragId % KT, mt = fragId / KT;
  int m = mt * 16 + (lane & 15);
  int k = kt * 32 + ((v >= 4) ? 16 : 0) + ((lane >= 16) ? 8 : 0) + 2 * (v & 3) + e;
  out[idx] = f2bf(x[(size_t)m * K + k]);
}

// ---------------------------------------------------------------------------
// Batched GEMM, 2x2 tile blocking per wave, K-loop fully unrolled at compile
// time (scheduler pipelines the global_load_b128s across WMMA groups).
// out[M,N] = A*B + bias (f32).  MT2 = MT/2, NT2 = NT/2.
// ---------------------------------------------------------------------------
template <int KT>
__global__ void __launch_bounds__(256) gemm_frag2x2_kernel(
    const unsigned short* __restrict__ afrag,
    const unsigned short* __restrict__ bfrag,
    const float* __restrict__ bias,
    float* __restrict__ out,
    int MT2, int NT2, int ldo) {
  int wid  = (blockIdx.x * blockDim.x + threadIdx.x) >> 5;
  int lane = threadIdx.x & 31;
  if (wid >= MT2 * NT2) return;                // wave-uniform exit
  int nt2 = wid % NT2, mt2 = wid / NT2;
  const v16bf* A0 = (const v16bf*)afrag + (size_t)(2 * mt2) * KT * 32 + lane;
  const v16bf* A1 = A0 + (size_t)KT * 32;
  const v16bf* B0 = (const v16bf*)bfrag + (size_t)(2 * nt2) * KT * 32 + lane;
  const v16bf* B1 = B0 + (size_t)KT * 32;
  v8f c00 = {}, c01 = {}, c10 = {}, c11 = {};
#pragma unroll
  for (int kt = 0; kt < KT; ++kt) {
    v16bf a0 = A0[kt * 32];
    v16bf a1 = A1[kt * 32];
    v16bf b0 = B0[kt * 32];
    v16bf b1 = B1[kt * 32];
    c00 = wmma_bf16(a0, b0, c00);
    c01 = wmma_bf16(a0, b1, c01);
    c10 = wmma_bf16(a1, b0, c10);
    c11 = wmma_bf16(a1, b1, c11);
  }
  int mbase = 2 * mt2 * 16 + ((lane >> 4) << 3);
#pragma unroll
  for (int ni = 0; ni < 2; ++ni) {
    int n = (2 * nt2 + ni) * 16 + (lane & 15);
    float bv = bias[n];
    v8f c0 = (ni == 0) ? c00 : c01;
    v8f c1 = (ni == 0) ? c10 : c11;
#pragma unroll
    for (int r = 0; r < 8; ++r) {
      out[(size_t)(mbase + r) * ldo + n] = c0[r] + bv;
      out[(size_t)(mbase + 16 + r) * ldo + n] = c1[r] + bv;
    }
  }
}

// ---------------------------------------------------------------------------
// Persistent GRU scan. 48 blocks x 8 wave32 = 384 waves = the 2x192 tiles of
// gh = h @ whh^T per step. Each block pins its 4 N-tiles of w_hh fragments
// (4*32 frags = 128KB) in LDS once; B reads are ds_load for all 512 steps.
// ---------------------------------------------------------------------------
__global__ void __launch_bounds__(256, 1) gru_scan_kernel(
    const float* __restrict__ gx,            // [B*T, G] (row = b*T + t)
    const unsigned short* __restrict__ whh,  // [NTG*KTH*512] B-frags
    const float* __restrict__ bhh,           // [G]
    float* __restrict__ h,                   // [B*H] f32 (init 0)
    unsigned short* __restrict__ hfrag,      // [MTB*KTH*512] A-frags of h (init 0)
    float* __restrict__ gh,                  // [B*G] f32 scratch
    unsigned short* __restrict__ seqfrag,    // out0 A-frags for next layer (or null)
    int* __restrict__ sync) {
  __shared__ unsigned short ldsw[LDS_NT * KTH * 512];   // 128 KB of 320 KB WGP LDS

  const int lane = threadIdx.x & 31;
  const int w    = threadIdx.x >> 5;             // 0..7 block-local wave
  const int gtid = blockIdx.x * blockDim.x + threadIdx.x;
  const int nthr = gridDim.x * blockDim.x;
  int* cnt = sync;
  int* gen = sync + 1;

  // ---- stage this block's w_hh fragment slice into LDS (once) ----
  {
    const unsigned short* src = whh + (size_t)(blockIdx.x * LDS_NT) * KTH * 512;
    for (int i = threadIdx.x * 8; i < LDS_NT * KTH * 512; i += blockDim.x * 8)
      *(uint4*)&ldsw[i] = *(const uint4*)&src[i];
  }
  __syncthreads();

  const int mt  = w & 1;                         // this wave's output tile
  const int lnt = w >> 1;                        // 0..3 local nt
  const int nt  = blockIdx.x * LDS_NT + lnt;
  const v16bf* A  = (const v16bf*)hfrag + (size_t)mt * KTH * 32 + lane;
  const v16bf* Bl = (const v16bf*)&ldsw[(size_t)lnt * KTH * 512] + lane;

  for (int t = 0; t < Tm; ++t) {
    // ---- phase 1: gh = h @ whh^T + bhh (A from cache, B from LDS) ----
    {
      v8f acc = {};
#pragma unroll
      for (int kt = 0; kt < KTH; ++kt) {
        v16bf a = A[kt * 32];
        v16bf b = Bl[kt * 32];
        acc = wmma_bf16(a, b, acc);
      }
      int n = nt * 16 + (lane & 15);
      float bv = bhh[n];
      int mbase = mt * 16 + ((lane >> 4) << 3);
#pragma unroll
      for (int r = 0; r < 8; ++r)
        gh[(size_t)(mbase + r) * Gm + n] = acc[r] + bv;
    }
    grid_sync(cnt, gen, SCAN_BLOCKS, 2 * t + 1);

    // ---- phase 2: gates + h update + repack fragments ----
    for (int i = gtid; i < Bm * Hm; i += nthr) {
      int b = i >> 10;
      int j = i & (Hm - 1);
      const float* gxr = gx + ((size_t)(b * Tm + t)) * Gm;
      const float* ghr = gh + (size_t)b * Gm;
      if (t + 1 < Tm) {                          // hide next step's gx HBM latency
        __builtin_prefetch(gxr + Gm + j, 0, 0);
        __builtin_prefetch(gxr + Gm + Hm + j, 0, 0);
        __builtin_prefetch(gxr + Gm + 2 * Hm + j, 0, 0);
      }
      float r = sigmoidf_(gxr[j] + ghr[j]);
      float z = sigmoidf_(gxr[Hm + j] + ghr[Hm + j]);
      float nn = tanhf(gxr[2 * Hm + j] + r * ghr[2 * Hm + j]);
      float hn = (1.0f - z) * nn + z * h[i];
      h[i] = hn;
      unsigned short hb = f2bf(hn);
      // inverse of A-frag layout for (row m, k = j)
      int m  = b & 15;
      int mtd = b >> 4;
      int kt = j >> 5;
      int kk = j & 31;
      int hi16 = (kk >> 4) & 1, lhi = (kk >> 3) & 1;
      int v = hi16 * 4 + ((kk >> 1) & 3), e = kk & 1;
      int lw = m + lhi * 16;
      hfrag[(size_t)(mtd * KTH + kt) * 512 + lw * 16 + v * 2 + e] = hb;
      if (seqfrag) {   // stream out0 as A-frags for layer-1 input projection
        int row = b * Tm + t;
        int ms = row & 15, mts = row >> 4;
        int ls = ms + lhi * 16;
        seqfrag[(size_t)(mts * KTH + kt) * 512 + ls * 16 + v * 2 + e] = hb;
      }
    }
    grid_sync(cnt, gen, SCAN_BLOCKS, 2 * t + 2);
  }
}

// ---------------------------------------------------------------------------
// FC epilogue: sigmoid(h_last1 @ fc_w^T + fc_b) -> d_out[0:B*OUT]
// ---------------------------------------------------------------------------
__global__ void __launch_bounds__(256) fc_kernel(
    const unsigned short* __restrict__ hfrag,  // final-h A-frags
    const unsigned short* __restrict__ fcw,    // fc B-frags
    const float* __restrict__ fcb,
    float* __restrict__ out) {
  int wid  = (blockIdx.x * blockDim.x + threadIdx.x) >> 5;
  int lane = threadIdx.x & 31;
  if (wid >= MTB * NTOUT) return;
  int nt = wid % NTOUT, mt = wid / NTOUT;
  const v16bf* A = (const v16bf*)hfrag + (size_t)mt * KTH * 32 + lane;
  const v16bf* Bp = (const v16bf*)fcw + (size_t)nt * KTH * 32 + lane;
  v8f acc = {};
#pragma unroll
  for (int kt = 0; kt < KTH; ++kt) {
    v16bf a = A[kt * 32];
    v16bf b = Bp[kt * 32];
    acc = wmma_bf16(a, b, acc);
  }
  int n = nt * 16 + (lane & 15);
  float bv = fcb[n];
  int mbase = mt * 16 + ((lane >> 4) << 3);
#pragma unroll
  for (int r = 0; r < 8; ++r)
    out[(size_t)(mbase + r) * OUTm + n] = sigmoidf_(acc[r] + bv);
}

// h_n = stack([h_last0, h_last1]) -> d_out[B*OUT : B*OUT + 2*B*H]
__global__ void copy_hn_kernel(const float* __restrict__ h0,
                               const float* __restrict__ h1,
                               float* __restrict__ out) {
  int i = blockIdx.x * blockDim.x + threadIdx.x;
  if (i < Bm * Hm)            out[Bm * OUTm + i] = h0[i];
  else if (i < 2 * Bm * Hm)   out[Bm * OUTm + i] = h1[i - Bm * Hm];
}

__global__ void init_scan_kernel(float* __restrict__ h,
                                 unsigned short* __restrict__ hfrag,
                                 int* __restrict__ sync) {
  int i = blockIdx.x * blockDim.x + threadIdx.x;
  if (i < Bm * Hm)  h[i] = 0.0f;
  if (i < MTB * KTH * 512) hfrag[i] = 0;
  if (i == 0) { sync[0] = 0; sync[1] = 0; }
}

// ---------------------------------------------------------------------------
extern "C" void kernel_launch(void* const* d_in, const int* in_sizes, int n_in,
                              void* d_out, int out_size, void* d_ws, size_t ws_size,
                              hipStream_t stream) {
  const float* input = (const float*)d_in[0];
  const float* w_ih0 = (const float*)d_in[1];
  const float* w_hh0 = (const float*)d_in[2];
  const float* b_ih0 = (const float*)d_in[3];
  const float* b_hh0 = (const float*)d_in[4];
  const float* w_ih1 = (const float*)d_in[5];
  const float* w_hh1 = (const float*)d_in[6];
  const float* b_ih1 = (const float*)d_in[7];
  const float* b_hh1 = (const float*)d_in[8];
  const float* fc_w  = (const float*)d_in[9];
  const float* fc_b  = (const float*)d_in[10];
  float* out = (float*)d_out;

  char* ws = (char*)d_ws;
  float*          gx    = (float*)(ws + OFF_GX);
  unsigned short* seq   = (unsigned short*)(ws + OFF_SEQ);
  unsigned short* wih0f = (unsigned short*)(ws + OFF_WIH0);
  unsigned short* whh0f = (unsigned short*)(ws + OFF_WHH0);
  unsigned short* wih1f = (unsigned short*)(ws + OFF_WIH1);
  unsigned short* whh1f = (unsigned short*)(ws + OFF_WHH1);
  unsigned short* fcwf  = (unsigned short*)(ws + OFF_FCW);
  float*          h0    = (float*)(ws + OFF_H0);
  float*          h1    = (float*)(ws + OFF_H1);
  unsigned short* hfrag = (unsigned short*)(ws + OFF_HFRAG);
  float*          gh    = (float*)(ws + OFF_GH);
  int*            sync  = (int*)(ws + OFF_SYNC);

  // ---- pack weights into WMMA fragment layout (bf16) ----
  pack_w_kernel<<<(NTG * KTI * 512) / 256, 256, 0, stream>>>(w_ih0, wih0f, NTG, KTI, INm);
  pack_w_kernel<<<(NTG * KTH * 512) / 256, 256, 0, stream>>>(w_hh0, whh0f, NTG, KTH, Hm);
  pack_w_kernel<<<(NTG * KTH * 512) / 256, 256, 0, stream>>>(w_ih1, wih1f, NTG, KTH, Hm);
  pack_w_kernel<<<(NTG * KTH * 512) / 256, 256, 0, stream>>>(w_hh1, whh1f, NTG, KTH, Hm);
  pack_w_kernel<<<(NTOUT * KTH * 512) / 256, 256, 0, stream>>>(fc_w, fcwf, NTOUT, KTH, Hm);

  // ---- layer 0: gx = x @ w_ih0^T + b_ih0 (batched over B*T) ----
  pack_a_kernel<<<(MTSEQ * KTI * 512) / 256, 256, 0, stream>>>(input, seq, MTSEQ, KTI, INm);
  gemm_frag2x2_kernel<KTI><<<((MTSEQ / 2) * (NTG / 2)) / 8, 256, 0, stream>>>(
      seq, wih0f, b_ih0, gx, MTSEQ / 2, NTG / 2, Gm);

  // ---- layer 0 scan (writes out0 A-frags into seq for layer 1) ----
  init_scan_kernel<<<(Bm * Hm) / 256, 256, 0, stream>>>(h0, hfrag, sync);
  gru_scan_kernel<<<SCAN_BLOCKS, 256, 0, stream>>>(gx, whh0f, b_hh0, h0, hfrag, gh, seq, sync);

  // ---- layer 1: gx = out0 @ w_ih1^T + b_ih1 ----
  gemm_frag2x2_kernel<KTH><<<((MTSEQ / 2) * (NTG / 2)) / 8, 256, 0, stream>>>(
      seq, wih1f, b_ih1, gx, MTSEQ / 2, NTG / 2, Gm);

  // ---- layer 1 scan ----
  init_scan_kernel<<<(Bm * Hm) / 256, 256, 0, stream>>>(h1, hfrag, sync);
  gru_scan_kernel<<<SCAN_BLOCKS, 256, 0, stream>>>(gx, whh1f, b_hh1, h1, hfrag, gh,
                                                   (unsigned short*)nullptr, sync);

  // ---- FC epilogue + h_n ----
  fc_kernel<<<(MTB * NTOUT) / 8, 256, 0, stream>>>(hfrag, fcwf, fc_b, out);
  copy_hn_kernel<<<(2 * Bm * Hm) / 256, 256, 0, stream>>>(h0, h1, out);
  (void)in_sizes; (void)n_in; (void)out_size; (void)ws_size;
}
